// MultiHeadAttention_20959440404973
// MI455X (gfx1250) — matrix operand order repaired and checked by reading, not compiled, not run on hardware
//
#include <hip/hip_runtime.h>
#include <hip/hip_bf16.h>

// ---------------- problem constants ----------------
#define D_MODEL   1024
#define NUM_HEADS 16
#define D_K       64
#define BATCH     4
#define SEQ       2048
#define M_TOK     (BATCH * SEQ)     // 8192 token rows
#define NKT       (D_MODEL / 32)    // 32 K-tiles of 32
#define NNT       (D_MODEL / 16)    // 64 N-tiles of 16

// ---------------- vector types ----------------
typedef __attribute__((ext_vector_type(16))) __bf16        v16bf;
typedef __attribute__((ext_vector_type(8)))  float         v8f;
typedef __attribute__((ext_vector_type(8)))  unsigned short u16x8;
typedef __attribute__((ext_vector_type(4)))  float         f32x4;

union frag_u { v16bf v; u16x8 h[2]; };

// Load a 16-bf16-per-lane fragment from two 16B-aligned halves.
__device__ inline v16bf ld_frag(const unsigned short* p0, const unsigned short* p1) {
  frag_u f;
  f.h[0] = *(const u16x8*)p0;
  f.h[1] = *(const u16x8*)p1;
  return f.v;
}

__device__ inline v8f vzero8() {
  v8f v;
#pragma unroll
  for (int i = 0; i < 8; ++i) v[i] = 0.0f;
  return v;
}

__device__ inline unsigned short bf16_bits(float x) {
  __bf16 t = (__bf16)x;
  return __builtin_bit_cast(unsigned short, t);
}

__device__ inline v8f wmma_bf16(v16bf a, v16bf b, v8f c) {
  // (neg_a, A, neg_b, B, c_mod, C, reuse_a, reuse_b)
  return __builtin_amdgcn_wmma_f32_16x16x32_bf16(false, a, false, b, (short)0, c,
                                                 false, false);
}

// Convert 16 fp32 (4x float4, in A-fragment element order) to a bf16 A-fragment.
__device__ inline v16bf cvt_frag(f32x4 a0, f32x4 a1, f32x4 a2, f32x4 a3) {
  v16bf v;
#pragma unroll
  for (int i = 0; i < 4; ++i) {
    v[i]      = (__bf16)a0[i];
    v[i + 4]  = (__bf16)a1[i];
    v[i + 8]  = (__bf16)a2[i];
    v[i + 12] = (__bf16)a3[i];
  }
  return v;
}

// CDNA5 async copy: one 16B element per lane, global -> LDS, ASYNCcnt-tracked.
__device__ inline void async_cp16(unsigned lds_off, unsigned long long gaddr) {
  asm volatile("global_load_async_to_lds_b128 %0, %1, off"
               :: "v"(lds_off), "v"(gaddr) : "memory");
}

// ------------------------------------------------------------------
// Repack a fp32 weight [K=1024][N=1024] (row-major) into bf16 WMMA
// B-fragment order: frag(kt,nt) holds the 32x16 tile; per lane 16 bf16
// (elements 0..7 -> K 8*half+e, elements 8..15 -> K 16+8*half+(e-8);
// N = lane%16).  Layout: [nt][kt][lane][16].
// ------------------------------------------------------------------
__global__ void pack_w(const float* __restrict__ W, unsigned short* __restrict__ P) {
  int idx = blockIdx.x * 256 + threadIdx.x;          // 1M elements
  int e     = idx & 15;
  int lane  = (idx >> 4) & 31;
  int frag  = idx >> 9;
  int kt    = frag & (NKT - 1);
  int nt    = frag >> 5;
  int n     = nt * 16 + (lane & 15);
  int k     = kt * 32 + (lane >> 4) * 8 + (e & 7) + (e >> 3) * 16;
  P[idx]    = bf16_bits(W[(size_t)k * D_MODEL + n]);
}

// ------------------------------------------------------------------
// Projection GEMM: C[8192x1024] = X(fp32) * Wp(packed bf16) + bias.
// One wave: 16x64 output strip (4 N-tiles), 32 K-steps of WMMA bf16.
// MODE 0: store bf16 [B,H,S,64]   (q / k)
// MODE 1: store bf16 [B,H,64,S]   (v transposed; packed b128 stores)
// ------------------------------------------------------------------
template <int MODE>
__global__ void proj_gemm(const float* __restrict__ X,
                          const unsigned short* __restrict__ Wp,
                          const float* __restrict__ bias,
                          unsigned short* __restrict__ outb) {
  const int lane  = threadIdx.x & 31;
  const int wid   = (blockIdx.x * blockDim.x + threadIdx.x) >> 5;
  const int mtile = wid >> 4;        // 512 M-tiles
  const int nt0   = (wid & 15) * 4;  // 4 consecutive N-tiles per wave
  const int l16 = lane & 15, half = lane >> 4, h8 = half * 8;
  const int arow = mtile * 16 + l16;

  v8f acc[4];
#pragma unroll
  for (int t = 0; t < 4; ++t) acc[t] = vzero8();

  for (int kb = 0; kb < D_MODEL; kb += 32) {
    const float* xp = X + (size_t)arow * D_MODEL + kb + h8;
    v16bf afrag = cvt_frag(*(const f32x4*)(xp),      *(const f32x4*)(xp + 4),
                           *(const f32x4*)(xp + 16), *(const f32x4*)(xp + 20));
    const int kt = kb >> 5;
#pragma unroll
    for (int t = 0; t < 4; ++t) {
      const unsigned short* bp =
          Wp + ((size_t)(((nt0 + t) * NKT + kt) * 32 + lane) << 4);
      acc[t] = wmma_bf16(afrag, ld_frag(bp, bp + 8), acc[t]);
    }
  }

#pragma unroll
  for (int t = 0; t < 4; ++t) {
    const int n  = (nt0 + t) * 16 + l16;
    const float bv = bias[n];
    const int hh = n >> 6, dd = n & 63;
    if (MODE == 0) {
#pragma unroll
      for (int r = 0; r < 8; ++r) {
        int m = mtile * 16 + r + 8 * half;
        int b = m >> 11, s = m & (SEQ - 1);
        outb[(((size_t)b * NUM_HEADS + hh) * SEQ + s) * D_K + dd] =
            bf16_bits(acc[t][r] + bv);
      }
    } else {
      u16x8 pk;
#pragma unroll
      for (int r = 0; r < 8; ++r) pk[r] = bf16_bits(acc[t][r] + bv);
      int m0 = mtile * 16 + 8 * half;          // 8 consecutive tokens
      int b = m0 >> 11, s0 = m0 & (SEQ - 1);
      *(u16x8*)(outb + (((size_t)b * NUM_HEADS + hh) * D_K + dd) * SEQ + s0) = pk;
    }
  }
}

// ------------------------------------------------------------------
// Causal flash attention, one wave per (b, h, 16-query tile).
// S^T = K * Q^T (rows = keys, cols = queries): softmax is a per-lane
// register reduction + one shfl_xor(16).  O^T = V^T * P^T.
// K and V^T blocks are double-buffered in LDS via CDNA5 async copies
// (global_load_async_to_lds_b128, ASYNCcnt), staged one block ahead.
// q,k: [B,H,S,64] bf16 ; vT: [B,H,64,S] bf16 ; out: [B*S, 1024] bf16.
// ------------------------------------------------------------------
__global__ void __launch_bounds__(64)
attn_fwd(const unsigned short* __restrict__ Qb,
         const unsigned short* __restrict__ Kb,
         const unsigned short* __restrict__ VT,
         unsigned short* __restrict__ Ob) {
  // per-wave: K 2x4KB, V 2x4KB, P 1KB  -> 17KB; 2 waves/block = 34KB
  __shared__ __align__(16) unsigned short kstage[2][2][32 * 64];
  __shared__ __align__(16) unsigned short vstage[2][2][64 * 32];
  __shared__ __align__(16) unsigned short ptile[2][512];

  const int lane  = threadIdx.x & 31;
  const int wib   = threadIdx.x >> 5;
  const int wid   = blockIdx.x * 2 + wib;
  const int qt    = wid & 127;   // S/16 query tiles
  const int bh    = wid >> 7;    // B*H = 64
  const int qbase = qt * 16;
  const int l16 = lane & 15, half = lane >> 4, h8 = half * 8;
  unsigned short* myP = ptile[wib];     // [query(16)][key(32)] bf16

  const unsigned short* qh = Qb + (size_t)bh * SEQ * D_K;
  const unsigned short* kh = Kb + (size_t)bh * SEQ * D_K;
  const unsigned short* vh = VT + (size_t)bh * D_K * SEQ;

  // LDS byte offsets of the stage buffers (low 32 bits of generic address).
  unsigned kOff[2], vOff[2];
  kOff[0] = (unsigned)(uintptr_t)&kstage[wib][0][0];
  kOff[1] = (unsigned)(uintptr_t)&kstage[wib][1][0];
  vOff[0] = (unsigned)(uintptr_t)&vstage[wib][0][0];
  vOff[1] = (unsigned)(uintptr_t)&vstage[wib][1][0];

  // Stage one 32-key block: K block is 4KB contiguous; V^T block is
  // 64 rows x 64B (row stride SEQ*2 bytes).  16 async b128 copies.
  auto stage_kv = [&](int buf, int keybase) {
    unsigned long long kg =
        (unsigned long long)(uintptr_t)(kh + (size_t)keybase * D_K) + (unsigned)lane * 16u;
    unsigned kl = kOff[buf] + lane * 16;
#pragma unroll
    for (int i = 0; i < 8; ++i)
      async_cp16(kl + i * 512, kg + (unsigned)i * 512u);
    unsigned long long vg =
        (unsigned long long)(uintptr_t)(vh + (size_t)(lane >> 2) * SEQ) +
        (size_t)keybase * 2 + (unsigned)(lane & 3) * 16u;
    unsigned vl = vOff[buf] + lane * 16;
#pragma unroll
    for (int i = 0; i < 8; ++i)
      async_cp16(vl + i * 512, vg + (unsigned long long)i * (8ull * SEQ * 2));
  };

  // Preload Q^T as two B-fragments (dims 0..31 and 32..63 of query l16).
  v16bf qbf[2];
  {
    const unsigned short* qp = qh + (size_t)(qbase + l16) * D_K + h8;
    qbf[0] = ld_frag(qp, qp + 16);
    qbf[1] = ld_frag(qp + 32, qp + 48);
  }

  // Per-lane softmax state for query (qbase + l16); both halves redundant.
  const int qglob = qbase + l16;
  float run_m = -3.0e38f, run_l = 0.0f;
  v8f o[4];
#pragma unroll
  for (int t = 0; t < 4; ++t) o[t] = vzero8();

  const int nblk = (qt >> 1) + 1;  // causal bound: only the LAST block is partial
  stage_kv(0, 0);                  // prologue: block 0 in flight

  for (int blk = 0; blk < nblk; ++blk) {
    const int cur = blk & 1;
    // Stage next block (other buffer) and retire current block's copies.
    if (blk + 1 < nblk) {
      asm volatile("s_wait_dscnt 0x0" ::: "memory");   // prior reads of that buf done
      stage_kv(cur ^ 1, (blk + 1) * 32);
      asm volatile("s_wait_asynccnt 0x10" ::: "memory"); // oldest 16 (= cur block) done
    } else {
      asm volatile("s_wait_asynccnt 0x0" ::: "memory");
    }
    const unsigned short* kc = &kstage[wib][cur][0];   // [key(32)][dim(64)]
    const unsigned short* vc = &vstage[wib][cur][0];   // [dim(64)][key(32)]

    // S^T = K * Q^T : two 16-key tiles (reg r = key r+8*half), K-dim 64.
    v8f s0 = vzero8(), s1 = vzero8();
    {
      const unsigned short* kp = kc + (size_t)l16 * D_K + h8;
      s0 = wmma_bf16(ld_frag(kp, kp + 16), qbf[0], s0);
      s0 = wmma_bf16(ld_frag(kp + 32, kp + 48), qbf[1], s0);
      const unsigned short* kp1 = kc + (size_t)(16 + l16) * D_K + h8;
      s1 = wmma_bf16(ld_frag(kp1, kp1 + 16), qbf[0], s1);
      s1 = wmma_bf16(ld_frag(kp1 + 32, kp1 + 48), qbf[1], s1);
    }

    // scale (+ causal mask only on the diagonal block)
    const int keybase = blk * 32;
#pragma unroll
    for (int r = 0; r < 8; ++r) { s0[r] *= 0.125f; s1[r] *= 0.125f; }
    if (blk == nblk - 1) {
#pragma unroll
      for (int r = 0; r < 8; ++r) {
        if (keybase + r + 8 * half > qglob)      s0[r] = -3.0e38f;
        if (keybase + 16 + r + 8 * half > qglob) s1[r] = -3.0e38f;
      }
    }

    // online softmax: per-lane register reduction + one cross-half shuffle
    float m = fmaxf(s0[0], s1[0]);
#pragma unroll
    for (int r = 1; r < 8; ++r) m = fmaxf(m, fmaxf(s0[r], s1[r]));
    m = fmaxf(m, __shfl_xor(m, 16));
    const float mnew  = fmaxf(run_m, m);
    const float alpha = __expf(run_m - mnew);
    run_m = mnew;
    float rs = 0.0f;
#pragma unroll
    for (int r = 0; r < 8; ++r) {
      s0[r] = __expf(s0[r] - mnew); rs += s0[r];
      s1[r] = __expf(s1[r] - mnew); rs += s1[r];
    }
    rs += __shfl_xor(rs, 16);
    run_l = run_l * alpha + rs;
#pragma unroll
    for (int t = 0; t < 4; ++t) {
#pragma unroll
      for (int r = 0; r < 8; ++r) o[t][r] *= alpha;
    }

    // P^T -> LDS [query][key] as two packed b128 stores per lane
    u16x8 pk0, pk1;
#pragma unroll
    for (int r = 0; r < 8; ++r) { pk0[r] = bf16_bits(s0[r]); pk1[r] = bf16_bits(s1[r]); }
    *(u16x8*)(myP + l16 * 32 + 8 * half)      = pk0;
    *(u16x8*)(myP + l16 * 32 + 16 + 8 * half) = pk1;
    asm volatile("s_wait_dscnt 0x0" ::: "memory");
    // P^T B-fragment: col = query l16, keys 8*half..+7 and 16+8*half..+7
    v16bf pb;
    {
      const unsigned short* lp = myP + l16 * 32 + h8;
      pb = ld_frag(lp, lp + 16);
    }

    // O^T += V^T * P^T : A-fragments from the staged V^T tile
#pragma unroll
    for (int t = 0; t < 4; ++t) {
      const unsigned short* vp = vc + (size_t)(t * 16 + l16) * 32 + h8;
      o[t] = wmma_bf16(ld_frag(vp, vp + 16), pb, o[t]);
    }
  }

  // normalize + store merged heads: O^T C-layout has 8 consecutive dims
  // per register column -> one b128 store per tile per lane.
  const int b = bh >> 4, hh = bh & 15;
  const float inv = 1.0f / run_l;
  const size_t row = (size_t)(b * SEQ + qglob) * D_MODEL;
#pragma unroll
  for (int t = 0; t < 4; ++t) {
    u16x8 pk;
#pragma unroll
    for (int r = 0; r < 8; ++r) pk[r] = bf16_bits(o[t][r] * inv);
    *(u16x8*)(Ob + row + hh * 64 + t * 16 + 8 * half) = pk;
  }
}

// ------------------------------------------------------------------
// Output projection: d_out(fp32) = A(bf16 [8192x1024]) * WoP + bo.
// ------------------------------------------------------------------
__global__ void out_gemm(const unsigned short* __restrict__ A,
                         const unsigned short* __restrict__ Wp,
                         const float* __restrict__ bias,
                         float* __restrict__ out) {
  const int lane  = threadIdx.x & 31;
  const int wid   = (blockIdx.x * blockDim.x + threadIdx.x) >> 5;
  const int mtile = wid >> 4;
  const int nt0   = (wid & 15) * 4;
  const int l16 = lane & 15, half = lane >> 4, h8 = half * 8;
  const int arow = mtile * 16 + l16;

  v8f acc[4];
#pragma unroll
  for (int t = 0; t < 4; ++t) acc[t] = vzero8();

  for (int kb = 0; kb < D_MODEL; kb += 32) {
    const unsigned short* ap = A + (size_t)arow * D_MODEL + kb + h8;
    v16bf afrag = ld_frag(ap, ap + 16);
    const int kt = kb >> 5;
#pragma unroll
    for (int t = 0; t < 4; ++t) {
      const unsigned short* bp =
          Wp + ((size_t)(((nt0 + t) * NKT + kt) * 32 + lane) << 4);
      acc[t] = wmma_bf16(afrag, ld_frag(bp, bp + 8), acc[t]);
    }
  }

#pragma unroll
  for (int t = 0; t < 4; ++t) {
    const int n = (nt0 + t) * 16 + l16;
    const float bv = bias[n];
#pragma unroll
    for (int r = 0; r < 8; ++r) {
      const int m = mtile * 16 + r + 8 * half;
      out[(size_t)m * D_MODEL + n] = acc[t][r] + bv;
    }
  }
}

// ------------------------------------------------------------------
extern "C" void kernel_launch(void* const* d_in, const int* in_sizes, int n_in,
                              void* d_out, int out_size, void* d_ws, size_t ws_size,
                              hipStream_t stream) {
  (void)in_sizes; (void)n_in; (void)out_size; (void)ws_size;
  const float* Q   = (const float*)d_in[0];
  const float* Kin = (const float*)d_in[1];
  const float* V   = (const float*)d_in[2];
  // d_in[3]: causal mask (handled analytically)
  const float* Wq = (const float*)d_in[4];
  const float* bq = (const float*)d_in[5];
  const float* Wk = (const float*)d_in[6];
  const float* bk = (const float*)d_in[7];
  const float* Wv = (const float*)d_in[8];
  const float* bv = (const float*)d_in[9];
  const float* Wo = (const float*)d_in[10];
  const float* bo = (const float*)d_in[11];
  float* out = (float*)d_out;

  // workspace carve-up (bf16 elements): 4x2MB packed W + 4x16MB activations = 72MB
  unsigned short* ws = (unsigned short*)d_ws;
  const size_t WSZ = (size_t)D_MODEL * D_MODEL;  // 1M elems
  const size_t TSZ = (size_t)M_TOK * D_MODEL;    // 8M elems
  unsigned short* WqP = ws;
  unsigned short* WkP = WqP + WSZ;
  unsigned short* WvP = WkP + WSZ;
  unsigned short* WoP = WvP + WSZ;
  unsigned short* qB  = WoP + WSZ;
  unsigned short* kB  = qB + TSZ;
  unsigned short* vT  = kB + TSZ;
  unsigned short* aB  = vT + TSZ;

  pack_w<<<4096, 256, 0, stream>>>(Wq, WqP);
  pack_w<<<4096, 256, 0, stream>>>(Wk, WkP);
  pack_w<<<4096, 256, 0, stream>>>(Wv, WvP);
  pack_w<<<4096, 256, 0, stream>>>(Wo, WoP);

  proj_gemm<0><<<2048, 128, 0, stream>>>(Q,   WqP, bq, qB);
  proj_gemm<0><<<2048, 128, 0, stream>>>(Kin, WkP, bk, kB);
  proj_gemm<1><<<2048, 128, 0, stream>>>(V,   WvP, bv, vT);

  attn_fwd<<<4096, 64, 0, stream>>>(qB, kB, vT, aB);

  out_gemm<<<2048, 128, 0, stream>>>(aB, WoP, bo, out);
}
